// Network_44684839747701
// MI455X (gfx1250) — matrix operand order repaired and checked by reading, not compiled, not run on hardware
//
#include <hip/hip_runtime.h>

typedef __attribute__((ext_vector_type(16))) _Float16 v16h;
typedef __attribute__((ext_vector_type(8)))  _Float16 f16x8;
typedef __attribute__((ext_vector_type(8)))  float    v8f;

#define NPTS    (4096 * 64)
#define TILE_P  64            // points per block
#define NT      (TILE_P / 16) // n-tiles per block = 4
#define PTJ     (256 / TILE_P)// threads per point in scalar phases = 4
#define SX      392           // bufX row stride in halves (>=384, 16B-aligned rows)
#define SH      264           // bufH row stride in halves (>=256, 16B-aligned rows)

// ---------------- weight f32 -> f16 convert with zero K-padding ----------------
__global__ void cvt_pad_f16(_Float16* __restrict__ dst, const float* __restrict__ src,
                            int rows, int K, int Kpad) {
  int idx = blockIdx.x * blockDim.x + threadIdx.x;
  if (idx >= rows * Kpad) return;
  int r = idx / Kpad, k = idx - r * Kpad;
  dst[idx] = (_Float16)(k < K ? src[r * K + k] : 0.0f);
}

// ---------------- one WMMA GEMM layer: dst[p][m] = act(W * src[p][:] + b) -------
// src/dst in LDS, [point][K] layout. Each wave owns an M-slice (MT tiles of 16)
// and iterates ALL NT n-tiles, so each A (weight) fragment is fetched from L2
// exactly once per block and reused from registers across the 4 n-tiles.
template <int KCH, int MT, bool RELU>
__device__ __forceinline__ void wmma_layer(const _Float16* __restrict__ Wh, int Kpad,
                                           const float* __restrict__ bias,
                                           const _Float16* src, int sstr,
                                           _Float16* dst, int dstr,
                                           int lane, int wave) {
  const int mBase = wave * 16 * MT;
  const int lh = lane & 15;
  const bool hi = lane >= 16;
  v8f acc[MT][NT] = {};
  for (int kc = 0; kc < KCH; ++kc) {
    const int k0 = kc << 5;
    // B fragments (32x16, K x points), one per n-tile
    v16h b[NT];
#pragma unroll
    for (int n = 0; n < NT; ++n) {
      const _Float16* bp = src + ((n << 4) + lh) * sstr + k0 + (hi ? 16 : 0);
      f16x8 blo = *(const f16x8*)bp;        // ds_load_b128
      f16x8 bhi = *(const f16x8*)(bp + 8);  // ds_load_b128
#pragma unroll
      for (int e = 0; e < 8; ++e) { b[n][e] = blo[e]; b[n][e + 8] = bhi[e]; }
    }
    // A fragments (16x32, M x K), one per m-tile; L2-resident weights
    v16h a[MT];
#pragma unroll
    for (int m = 0; m < MT; ++m) {
      const _Float16* ap = Wh + (mBase + (m << 4) + lh) * Kpad + k0 + (hi ? 16 : 0);
      f16x8 alo = *(const f16x8*)ap;        // global_load_b128
      f16x8 ahi = *(const f16x8*)(ap + 8);  // global_load_b128
#pragma unroll
      for (int e = 0; e < 8; ++e) { a[m][e] = alo[e]; a[m][e + 8] = ahi[e]; }
    }
#pragma unroll
    for (int m = 0; m < MT; ++m)
#pragma unroll
      for (int n = 0; n < NT; ++n)
        acc[m][n] = __builtin_amdgcn_wmma_f32_16x16x32_f16(false, a[m], false, b[n],
                                                           (short)0, acc[m][n],
                                                           false, false);
  }
  __syncthreads();  // all reads of src done before anyone overwrites (Wl: dst==src)
#pragma unroll
  for (int m = 0; m < MT; ++m) {
    const int mO = mBase + (m << 4) + (hi ? 8 : 0);
#pragma unroll
    for (int n = 0; n < NT; ++n) {
      f16x8 o;
#pragma unroll
      for (int r = 0; r < 8; ++r) {
        float v = acc[m][n][r] + bias[mO + r];
        if (RELU) v = fmaxf(v, 0.0f);
        o[r] = (_Float16)v;
      }
      *(f16x8*)(dst + ((n << 4) + lh) * dstr + mO) = o;  // ds_store_b128
    }
  }
  __syncthreads();
}

// ---------------- trilinear gather of C channels from one volume ----------------
__device__ __forceinline__ void gather_vol(const float* __restrict__ vol,
                                           int C, int D, int H, int Wd,
                                           float gx, float gy, float gz,
                                           _Float16* dstRow, int kBase, int j) {
  float ix = (gx + 1.0f) * 0.5f * (float)(Wd - 1);
  float iy = (gy + 1.0f) * 0.5f * (float)(H - 1);
  float iz = (gz + 1.0f) * 0.5f * (float)(D - 1);
  float x0f = floorf(ix), y0f = floorf(iy), z0f = floorf(iz);
  float wx = ix - x0f, wy = iy - y0f, wz = iz - z0f;
  int x0 = (int)x0f, y0 = (int)y0f, z0 = (int)z0f;
  int   xi[2], yi[2], zi[2];
  float wxv[2] = {1.0f - wx, wx}, wyv[2] = {1.0f - wy, wy}, wzv[2] = {1.0f - wz, wz};
  bool  vx[2], vy[2], vz[2];
#pragma unroll
  for (int d = 0; d < 2; ++d) {
    int xc = x0 + d, yc = y0 + d, zc = z0 + d;
    vx[d] = (xc >= 0) && (xc < Wd);  xi[d] = min(max(xc, 0), Wd - 1);
    vy[d] = (yc >= 0) && (yc < H);   yi[d] = min(max(yc, 0), H - 1);
    vz[d] = (zc >= 0) && (zc < D);   zi[d] = min(max(zc, 0), D - 1);
  }
  float w8[8]; int off[8];
#pragma unroll
  for (int dz = 0; dz < 2; ++dz)
#pragma unroll
    for (int dy = 0; dy < 2; ++dy)
#pragma unroll
      for (int dx = 0; dx < 2; ++dx) {
        int t = (dz * 2 + dy) * 2 + dx;
        w8[t]  = (vx[dx] && vy[dy] && vz[dz]) ? wxv[dx] * wyv[dy] * wzv[dz] : 0.0f;
        off[t] = (zi[dz] * H + yi[dy]) * Wd + xi[dx];
      }
  const int DHW = D * H * Wd;
  for (int c = j; c < C; c += PTJ) {
    const float* vc = vol + c * DHW;
    float acc = 0.0f;
#pragma unroll
    for (int t = 0; t < 8; ++t) acc += w8[t] * vc[off[t]];
    dstRow[kBase + c] = (_Float16)acc;
  }
}

// ---------------- fused NeRF network: 64 points / block, 8 wave32s --------------
__global__ __launch_bounds__(256) void nerf_fused(
    const float* __restrict__ pts, const float* __restrict__ viewdir,
    const float* __restrict__ vol1, const float* __restrict__ vol2,
    const float* __restrict__ vol3, const float* __restrict__ vol4,
    const float* __restrict__ bounds, const int* __restrict__ latent_index,
    const float* __restrict__ latent_table,
    const float* __restrict__ b0, const float* __restrict__ b1,
    const float* __restrict__ b2, const float* __restrict__ Wa,
    const float* __restrict__ ba, const float* __restrict__ bf,
    const float* __restrict__ bl, const float* __restrict__ bv,
    const float* __restrict__ Wr, const float* __restrict__ br,
    const _Float16* __restrict__ W0h, const _Float16* __restrict__ W1h,
    const _Float16* __restrict__ W2h, const _Float16* __restrict__ Wfh,
    const _Float16* __restrict__ Wlh, const _Float16* __restrict__ Wvh,
    float* __restrict__ out) {
  __shared__ _Float16 bufX[TILE_P * SX];  // feats / 384-wide concat inputs (~50KB)
  __shared__ _Float16 bufH[TILE_P * SH];  // 256/128-wide activations (~34KB)
  __shared__ _Float16 vep[TILE_P * 96];   // [ve(27) | pe(63) | zeros(6)] (~12KB)

  const int tid  = threadIdx.x;
  const int lane = tid & 31, wave = tid >> 5;
  const int p = tid / PTJ, j = tid % PTJ; // PTJ threads per point
  const int i = blockIdx.x * TILE_P + p;  // global point index

  // ---- stage 0: grid-sample feats (K=0..351) + posenc scratch ----
  const float px = pts[i * 3 + 0], py = pts[i * 3 + 1], pz = pts[i * 3 + 2];
  const float gx = 2.0f * (px - bounds[2]) / (bounds[5] - bounds[2]) - 1.0f;
  const float gy = 2.0f * (py - bounds[1]) / (bounds[4] - bounds[1]) - 1.0f;
  const float gz = 2.0f * (pz - bounds[0]) / (bounds[3] - bounds[0]) - 1.0f;
  _Float16* row = &bufX[p * SX];
  gather_vol(vol1,  32, 96, 96, 96, gx, gy, gz, row,   0, j);
  gather_vol(vol2,  64, 48, 48, 48, gx, gy, gz, row,  32, j);
  gather_vol(vol3, 128, 24, 24, 24, gx, gy, gz, row,  96, j);
  gather_vol(vol4, 128, 12, 12, 12, gx, gy, gz, row, 224, j);

  const float vd[3] = {viewdir[i * 3], viewdir[i * 3 + 1], viewdir[i * 3 + 2]};
  const float pp[3] = {px, py, pz};
  for (int idx = j; idx < 96; idx += PTJ) {
    float v = 0.0f;
    if (idx < 3) v = vd[idx];
    else if (idx < 27) {  // posenc(viewdir, L=4)
      int r = idx - 3, l = r / 6, rm = r % 6, s = rm / 3, c = rm % 3;
      float a = (float)(1 << l) * vd[c];
      v = s ? cosf(a) : sinf(a);
    } else if (idx < 30) v = pp[idx - 27];
    else if (idx < 90) {  // posenc(pts, L=10)
      int r = idx - 30, l = r / 6, rm = r % 6, s = rm / 3, c = rm % 3;
      float a = (float)(1 << l) * pp[c];
      v = s ? cosf(a) : sinf(a);
    }
    vep[p * 96 + idx] = (_Float16)v;
  }
  __syncthreads();

  // ---- h0 = relu(W0*feats + b0)  [K=352] ----
  wmma_layer<11, 2, true>(W0h, 384, b0, bufX, SX, bufH, SH, lane, wave);

  // latent rows 256..383 of bufX (read later by Wl; rows 0..255 get h1/hf)
  {
    int li = latent_index[i];
    for (int c = j; c < 128; c += PTJ)
      bufX[p * SX + 256 + c] = (_Float16)latent_table[li * 128 + c];
  }
  __syncthreads();

  wmma_layer<8, 2, true>(W1h, 256, b1, bufH, SH, bufX, SX, lane, wave);   // h1
  wmma_layer<8, 2, true>(W2h, 256, b2, bufX, SX, bufH, SH, lane, wave);   // h2

  // sigma head: 1x256 dot per point
  if (tid < TILE_P) {
    float s = ba[0];
    for (int c = 0; c < 256; ++c) s += Wa[c] * (float)bufH[tid * SH + c];
    out[blockIdx.x * TILE_P + tid] = s;
  }

  wmma_layer<8, 2, false>(Wfh, 256, bf, bufH, SH, bufX, SX, lane, wave);  // hf -> rows 0..255
  wmma_layer<12, 2, false>(Wlh, 384, bl, bufX, SX, bufX, SX, lane, wave); // hl = Wl*[hf|lat]

  // splice [ve|pe|0] into rows 256..351 for the Wv concat (K=346, padded 352)
  for (int r = j; r < 96; r += PTJ) bufX[p * SX + 256 + r] = vep[p * 96 + r];
  __syncthreads();

  wmma_layer<11, 1, true>(Wvh, 384, bv, bufX, SX, bufH, SH, lane, wave);  // 128-wide

  // rgb head: 3x128 dot per point
  if (tid < TILE_P * 3) {
    int pl = tid % TILE_P, c = tid / TILE_P;
    int gi = blockIdx.x * TILE_P + pl;
    float s = br[c];
    for (int k = 0; k < 128; ++k) s += Wr[c * 128 + k] * (float)bufH[pl * SH + k];
    out[NPTS + (gi >> 6) * 192 + c * 64 + (gi & 63)] = s;
  }
}

extern "C" void kernel_launch(void* const* d_in, const int* in_sizes, int n_in,
                              void* d_out, int out_size, void* d_ws, size_t ws_size,
                              hipStream_t stream) {
  (void)in_sizes; (void)n_in; (void)out_size; (void)ws_size;
  const float* pts    = (const float*)d_in[0];
  const float* vdir   = (const float*)d_in[1];
  const float* vol1   = (const float*)d_in[2];
  const float* vol2   = (const float*)d_in[3];
  const float* vol3   = (const float*)d_in[4];
  const float* vol4   = (const float*)d_in[5];
  const float* bounds = (const float*)d_in[6];
  const int*   lidx   = (const int*)d_in[7];
  const float* ltab   = (const float*)d_in[8];
  const float* W0 = (const float*)d_in[9],  *b0 = (const float*)d_in[10];
  const float* W1 = (const float*)d_in[11], *b1 = (const float*)d_in[12];
  const float* W2 = (const float*)d_in[13], *b2 = (const float*)d_in[14];
  const float* Wa = (const float*)d_in[15], *ba = (const float*)d_in[16];
  const float* Wf = (const float*)d_in[17], *bf = (const float*)d_in[18];
  const float* Wl = (const float*)d_in[19], *bl = (const float*)d_in[20];
  const float* Wv = (const float*)d_in[21], *bv = (const float*)d_in[22];
  const float* Wr = (const float*)d_in[23], *br = (const float*)d_in[24];
  float* out = (float*)d_out;

  // f16 weight workspace layout (element offsets)
  _Float16* ws = (_Float16*)d_ws;
  _Float16* W0h = ws;                  // 256 x 384
  _Float16* W1h = W0h + 256 * 384;     // 256 x 256
  _Float16* W2h = W1h + 256 * 256;     // 256 x 256
  _Float16* Wfh = W2h + 256 * 256;     // 256 x 256
  _Float16* Wlh = Wfh + 256 * 256;     // 256 x 384
  _Float16* Wvh = Wlh + 256 * 384;     // 128 x 384

  auto cvt = [&](_Float16* dst, const float* src, int rows, int K, int Kpad) {
    int total = rows * Kpad;
    cvt_pad_f16<<<(total + 255) / 256, 256, 0, stream>>>(dst, src, rows, K, Kpad);
  };
  cvt(W0h, W0, 256, 352, 384);
  cvt(W1h, W1, 256, 256, 256);
  cvt(W2h, W2, 256, 256, 256);
  cvt(Wfh, Wf, 256, 256, 256);
  cvt(Wlh, Wl, 256, 384, 384);
  cvt(Wvh, Wv, 128, 346, 384);

  nerf_fused<<<NPTS / TILE_P, 256, 0, stream>>>(
      pts, vdir, vol1, vol2, vol3, vol4, bounds, lidx, ltab,
      b0, b1, b2, Wa, ba, bf, bl, bv, Wr, br,
      W0h, W1h, W2h, Wfh, Wlh, Wvh, out);
}
